// downsampleBlock_65687229826152
// MI455X (gfx1250) — compile-verified, matrix operand order checked by reading, not run on hardware
//
#include <hip/hip_runtime.h>
#include <hip/hip_bf16.h>

#define B_    16
#define N_    4096
#define S_    1024
#define K_    8
#define CIN_  128
#define COUT_ 256
#define R2_   (0.2f * 0.2f)
#define BN_EPS_ 1e-5f
#define WPITCH 136   // halves per LDS W row: 128 + 8 pad (272B => bank stride 4)

typedef _Float16 v16h __attribute__((ext_vector_type(16)));
typedef _Float16 v8h  __attribute__((ext_vector_type(8)));
typedef float    v8f  __attribute__((ext_vector_type(8)));

// ---------------- Kernel 1: farthest point sampling (one block per batch) ---------------
// 1024 threads = 32 wave32's. Argmax per iteration: 5-step intra-wave butterfly (no
// barriers), 32 wave leaders -> LDS, wave 0 does a second 32-lane butterfly. 2 barriers
// per iteration instead of 9.
__global__ __launch_bounds__(1024) void fps_kernel(const float* __restrict__ xyz,
                                                   int* __restrict__ fps_idx,
                                                   float* __restrict__ new_xyz) {
    __shared__ float lx[N_], ly[N_], lz[N_];
    __shared__ float rval[32];
    __shared__ int   ridx[32];
    __shared__ int   scur;
    const int b    = blockIdx.x;
    const int tid  = threadIdx.x;
    const int lane = tid & 31;
    const int wv   = tid >> 5;
    const float* base = xyz + (size_t)b * 3 * N_;

    float px[4], py[4], pz[4], dst[4];
#pragma unroll
    for (int i = 0; i < 4; ++i) {
        int n = i * 1024 + tid;
        px[i] = base[n]; py[i] = base[N_ + n]; pz[i] = base[2 * N_ + n];
        lx[n] = px[i]; ly[n] = py[i]; lz[n] = pz[i];
        dst[i] = 3.4e38f;
    }
    if (tid == 0) scur = 0;
    __syncthreads();

    for (int s = 0; s < S_; ++s) {
        int cur = scur;
        float cx = lx[cur], cy = ly[cur], cz = lz[cur];
        if (tid == 0) {
            fps_idx[b * S_ + s] = cur;
            float* o = new_xyz + ((size_t)b * S_ + s) * 3;
            o[0] = cx; o[1] = cy; o[2] = cz;
        }
        float bv = -1.0f; int bi = 0;
#pragma unroll
        for (int i = 0; i < 4; ++i) {
            float dx = px[i] - cx, dy = py[i] - cy, dz = pz[i] - cz;
            float d  = dx * dx + dy * dy + dz * dz;
            float nd = fminf(dst[i], d);
            dst[i] = nd;
            int n = i * 1024 + tid;
            if (nd > bv) { bv = nd; bi = n; }   // strict > keeps lowest index in-thread
        }
        // intra-wave argmax butterfly (lowest index wins ties -> matches jnp.argmax)
#pragma unroll
        for (int off = 16; off > 0; off >>= 1) {
            float ov = __shfl_xor(bv, off, 32);
            int   oi = __shfl_xor(bi, off, 32);
            if (ov > bv || (ov == bv && oi < bi)) { bv = ov; bi = oi; }
        }
        if (lane == 0) { rval[wv] = bv; ridx[wv] = bi; }
        __syncthreads();
        if (wv == 0) {
            float v2 = rval[lane]; int i2 = ridx[lane];
#pragma unroll
            for (int off = 16; off > 0; off >>= 1) {
                float ov = __shfl_xor(v2, off, 32);
                int   oi = __shfl_xor(i2, off, 32);
                if (ov > v2 || (ov == v2 && oi < i2)) { v2 = ov; i2 = oi; }
            }
            if (lane == 0) scur = i2;
        }
        __syncthreads();
    }
}

// ---------------- Kernel 2: ball query, wave-cooperative first-K-by-index ---------------
// One wave per centroid scan: 32 lanes test 32 candidate points per step; __ballot gives
// a wave-uniform in-ball mask that is walked scalar-uniformly (first K by index), with
// early exit. Block = 256 threads = 8 waves; each wave covers 128 centroids.
__global__ __launch_bounds__(256) void ballq_kernel(const float* __restrict__ xyz,
                                                    const int* __restrict__ fps_idx,
                                                    int* __restrict__ ball_idx) {
    __shared__ float lx[N_], ly[N_], lz[N_];
    const int b    = blockIdx.x;
    const int tid  = threadIdx.x;
    const int lane = tid & 31;
    const int wv   = tid >> 5;
    const float* base = xyz + (size_t)b * 3 * N_;
#pragma unroll
    for (int i = 0; i < 16; ++i) {
        int n = i * 256 + tid;
        lx[n] = base[n]; ly[n] = base[N_ + n]; lz[n] = base[2 * N_ + n];
    }
    __syncthreads();

    for (int it = 0; it < S_ / 8; ++it) {
        int s  = it * 8 + wv;
        int ci = fps_idx[b * S_ + s];
        float cx = lx[ci], cy = ly[ci], cz = lz[ci];
        int myidx = 0;   // lane k ends up holding the k-th in-ball index
        int cnt   = 0;   // wave-uniform
        for (int bb = 0; bb < N_ && cnt < K_; bb += 32) {
            int n = bb + lane;
            float dx = lx[n] - cx, dy = ly[n] - cy, dz = lz[n] - cz;
            bool in = (dx * dx + dy * dy + dz * dz) <= R2_;
            unsigned mm = (unsigned)__ballot(in);    // wave32: low 32 bits
            while (mm && cnt < K_) {
                int l = __builtin_ctz(mm);
                if (cnt == lane) myidx = bb + l;
                ++cnt;
                mm &= mm - 1;
            }
        }
        int first = __shfl(myidx, 0, 32);            // centroid itself guarantees cnt>=1
        if (lane >= cnt) myidx = first;              // pad with first in-ball index
        if (lane < K_) ball_idx[((size_t)b * S_ + s) * K_ + lane] = myidx;
    }
}

// ------ Kernel 3: gather + WMMA pointwise conv + BN + ReLU + max over K -----------------
// Block = 128 threads (4 waves). Each wave handles 2 centroids per pass (16 A-rows =
// 2 centroids x 8 neighbors). 64 centroids per block, COUT processed in two halves of 128.
__global__ __launch_bounds__(128) void sa_gemm_kernel(const float* __restrict__ feat,
                                                      const float* __restrict__ conv_w,
                                                      const float* __restrict__ conv_b,
                                                      const float* __restrict__ gamma,
                                                      const float* __restrict__ beta,
                                                      const float* __restrict__ mean,
                                                      const float* __restrict__ var,
                                                      const int* __restrict__ ball_idx,
                                                      float* __restrict__ out_feat) {
    __shared__ _Float16 wl[128 * WPITCH];   // one COUT-half of W, f16, padded rows
    __shared__ float ssc[128], sbi[128];    // fused BN scale / bias per channel
    const int b     = blockIdx.x;
    const int chunk = blockIdx.y;
    const int tid   = threadIdx.x;
    const int w     = tid >> 5;     // wave id 0..3
    const int lane  = tid & 31;
    const int m     = lane & 15;    // A-row / D-column index
    const int kh    = lane >> 4;    // K-half selector

    for (int oh = 0; oh < 2; ++oh) {
        // Stage this half of W into LDS as f16 (rows oh*128 .. oh*128+127).
        for (int idx = tid; idx < 128 * CIN_; idx += 128) {
            int o = idx >> 7, c = idx & 127;
            wl[o * WPITCH + c] = (_Float16)conv_w[(size_t)(oh * 128 + o) * CIN_ + c];
        }
        {   // fused BN affine: y = sc*dot + bi, with conv bias folded in
            int o = tid;
            if (o < 128) {
                int og = oh * 128 + o;
                float sc = gamma[og] * rsqrtf(var[og] + BN_EPS_);
                ssc[o] = sc;
                sbi[o] = beta[og] + sc * (conv_b[og] - mean[og]);
            }
        }
        __syncthreads();

        for (int p = 0; p < 8; ++p) {
            int s0 = chunk * 64 + (p * 4 + w) * 2;     // centroid pair (s0, s0+1)
            int sm = s0 + (m >> 3);                    // this lane's A-row centroid
            int kk = m & 7;                            // neighbor slot
            int n  = ball_idx[((size_t)b * S_ + sm) * K_ + kk];
            const float* fb = feat + (size_t)b * CIN_ * N_ + n;

            // Build A fragments: lane holds row m, K-halves per CDNA5 16-bit A layout.
            v16h af[4];
#pragma unroll
            for (int ks = 0; ks < 4; ++ks) {
                int c0 = ks * 32 + kh * 8;
#pragma unroll
                for (int i = 0; i < 8; ++i) {
                    af[ks][i]     = (_Float16)fb[(size_t)(c0 + i) * N_];
                    af[ks][i + 8] = (_Float16)fb[(size_t)(c0 + 16 + i) * N_];
                }
            }

#pragma unroll
            for (int nt = 0; nt < 8; ++nt) {
                int ocol = nt * 16 + m;                // channel within this COUT-half
                const _Float16* wrow = &wl[ocol * WPITCH];
                v8f acc = {};
#pragma unroll
                for (int ks = 0; ks < 4; ++ks) {
                    int c0 = ks * 32 + kh * 8;
                    v8h b0 = *(const v8h*)(wrow + c0);        // ds_load_b128
                    v8h b1 = *(const v8h*)(wrow + c0 + 16);   // ds_load_b128
                    v16h bf;
#pragma unroll
                    for (int i = 0; i < 8; ++i) { bf[i] = b0[i]; bf[i + 8] = b1[i]; }
                    acc = __builtin_amdgcn_wmma_f32_16x16x32_f16(
                        false, af[ks], false, bf, (short)0, acc, false, false);
                }
                // D layout: lanes 0-15 hold centroid s0's 8 K-rows, lanes 16-31 s0+1's.
                // BN + ReLU + max over K is lane-local.
                float sc = ssc[ocol], bi = sbi[ocol];
                float mx = 0.0f;                        // relu floor
#pragma unroll
                for (int v = 0; v < 8; ++v) {
                    float x = acc[v] * sc + bi;
                    mx = fmaxf(mx, fmaxf(x, 0.0f));
                }
                int sx = s0 + kh;
                int oc = oh * 128 + nt * 16 + m;
                out_feat[((size_t)b * S_ + sx) * COUT_ + oc] = mx;
            }
        }
        __syncthreads();
    }
}

extern "C" void kernel_launch(void* const* d_in, const int* in_sizes, int n_in,
                              void* d_out, int out_size, void* d_ws, size_t ws_size,
                              hipStream_t stream) {
    const float* xyz  = (const float*)d_in[0];
    const float* feat = (const float*)d_in[1];
    const float* cw   = (const float*)d_in[2];
    const float* cb   = (const float*)d_in[3];
    const float* gam  = (const float*)d_in[4];
    const float* bet  = (const float*)d_in[5];
    const float* mean = (const float*)d_in[6];
    const float* var  = (const float*)d_in[7];

    float* out      = (float*)d_out;
    float* new_xyz  = out;                           // [B,S,3]
    float* new_feat = out + (size_t)B_ * S_ * 3;     // [B,S,COUT]

    int* fps_idx  = (int*)d_ws;                      // B*S ints
    int* ball_idx = fps_idx + B_ * S_;               // B*S*K ints

    fps_kernel<<<B_, 1024, 0, stream>>>(xyz, fps_idx, new_xyz);
    ballq_kernel<<<B_, 256, 0, stream>>>(xyz, fps_idx, ball_idx);
    sa_gemm_kernel<<<dim3(B_, S_ / 64), 128, 0, stream>>>(
        feat, cw, cb, gam, bet, mean, var, ball_idx, new_feat);
}
